// VoiceCloneModel_25984552141471
// MI455X (gfx1250) — compile-verified, matrix operand order checked by reading, not run on hardware
//
#include <hip/hip_runtime.h>
#include <hip/hip_bf16.h>
#include <cstdint>

// ---------------------------------------------------------------------------
// Model dims
// ---------------------------------------------------------------------------
#define Bn    32
#define TT    512      // text length
#define TM    1000     // mel length
#define Hn    256      // hidden
#define En    512      // 2H
#define Gn    1024     // 4H gates
#define MELn  80
#define NHEAD 4
#define HD    128      // E / NHEAD

typedef __attribute__((ext_vector_type(16))) __bf16 v16bf;
typedef __attribute__((ext_vector_type(8)))  float  v8f;

union ABfrag { v16bf v; unsigned u[8]; };

__device__ __forceinline__ float sigf(float x) {
    return 1.0f / (1.0f + __expf(-x));
}

// K-pair offset for 16-bit WMMA A/B fragment VGPR v (per CDNA5 ISA 7.12.2):
// lanes 0-15 hold K pairs {0,1},{2,3},{4,5},{6,7} then {16..23};
// lanes 16-31 hold {8..15} then {24..31}.
__device__ __forceinline__ int frag_koff(int v, int half) {
    return (v < 4) ? (half * 8 + 2 * v) : (16 + half * 8 + 2 * (v - 4));
}

// ---------------------------------------------------------------------------
// f32 -> bf16 conversion (grid-stride)
// ---------------------------------------------------------------------------
__global__ void f32_to_bf16_kernel(const float* __restrict__ src,
                                   __bf16* __restrict__ dst, size_t n) {
    size_t i = (size_t)blockIdx.x * blockDim.x + threadIdx.x;
    size_t stride = (size_t)gridDim.x * blockDim.x;
    for (; i < n; i += stride) dst[i] = (__bf16)src[i];
}

// ---------------------------------------------------------------------------
// Embedding gather -> bf16 activations  x_bf[b*TT+t, H]
// ---------------------------------------------------------------------------
__global__ void embed_kernel(const int* __restrict__ text,
                             const float* __restrict__ emb,
                             __bf16* __restrict__ xbf) {
    const int total = Bn * TT * Hn;
    int i = blockIdx.x * blockDim.x + threadIdx.x;
    int stride = gridDim.x * blockDim.x;
    for (; i < total; i += stride) {
        int row = i >> 8;          // b*TT + t
        int col = i & (Hn - 1);
        int tok = text[row];
        xbf[i] = (__bf16)emb[tok * Hn + col];
    }
}

// ---------------------------------------------------------------------------
// Generic WMMA GEMM:  C[M,N] = A[M,K](bf16,row-major) * W[N,K]^T(bf16) + bias
// One 16x16 C-tile per wave, 4 waves / block. M,N multiples of 16, K of 32.
// ---------------------------------------------------------------------------
__global__ void gemm_bf16_wmma(const __bf16* __restrict__ A, int lda,
                               const __bf16* __restrict__ W, int ldw,
                               float* __restrict__ Cf,
                               __bf16* __restrict__ Cbf, int ldc,
                               const float* __restrict__ bias0,
                               const float* __restrict__ bias1,
                               int M, int N, int K) {
    const int lane = threadIdx.x & 31;
    const int wave = threadIdx.x >> 5;
    const int ntN = N >> 4;
    const int tile = blockIdx.x * (blockDim.x >> 5) + wave;
    if (tile >= (M >> 4) * ntN) return;
    const int mt = tile / ntN, nt = tile % ntN;
    const int m0 = mt << 4, n0 = nt << 4;
    const int ln = lane & 15, half = lane >> 4;

    const size_t arow = (size_t)(m0 + ln) * lda;
    const size_t brow = (size_t)(n0 + ln) * ldw;

    v8f acc = {};
    for (int k0 = 0; k0 < K; k0 += 32) {
        ABfrag a, b;
#pragma unroll
        for (int v = 0; v < 8; ++v) {
            const int kb = k0 + frag_koff(v, half);
            a.u[v] = *(const unsigned*)(A + arow + kb);
            b.u[v] = *(const unsigned*)(W + brow + kb);
        }
        if (k0 + 32 < K) {
            __builtin_prefetch(A + arow + k0 + 32, 0, 1);  // global_prefetch_b8
            __builtin_prefetch(W + brow + k0 + 32, 0, 1);
        }
        acc = __builtin_amdgcn_wmma_f32_16x16x32_bf16(
            false, a.v, false, b.v, (short)0, acc, false, false);
    }

    const int n = n0 + ln;
    float bias = 0.f;
    if (bias0) bias += bias0[n];
    if (bias1) bias += bias1[n];
#pragma unroll
    for (int r = 0; r < 8; ++r) {
        const int m = m0 + half * 8 + r;
        const float val = acc[r] + bias;
        if (Cf)  Cf[(size_t)m * ldc + n] = val;
        if (Cbf) Cbf[(size_t)m * ldc + n] = (__bf16)val;
    }
}

// ---------------------------------------------------------------------------
// Persistent LSTM recurrence (one workgroup of 1024 threads = 32 waves).
//   gates[32,1024] = xg[t] + h @ Whh^T   (WMMA, h in LDS bf16, Whh L2-resident)
// Wave w (w<16) owns gate N-tiles {w, w+16, w+32, w+48} for M-tile (w>>4):
// all 4 gates of hidden unit j land in the SAME lane -> activations are pure
// per-lane VALU, c-state lives in registers across all T steps.
// blockIdx.x selects param set (fwd/bwd encoder in one launch).
// ---------------------------------------------------------------------------
__global__ void lstm_rec_wmma(int T, int xg_has_t,
                              const float* __restrict__ xg0,
                              const float* __restrict__ xg1,
                              const __bf16* __restrict__ whh0,
                              const __bf16* __restrict__ whh1,
                              __bf16* __restrict__ out0,
                              __bf16* __restrict__ out1,
                              int out_ld, int col0a, int col0b,
                              int rev0, int rev1) {
    __shared__ __bf16 hbf[Bn * Hn];   // 16 KB: h state as bf16

    const float*  xg  = blockIdx.x ? xg1  : xg0;
    const __bf16* whh = blockIdx.x ? whh1 : whh0;
    __bf16*       out = blockIdx.x ? out1 : out0;
    const int col0 = blockIdx.x ? col0b : col0a;
    const int rev  = blockIdx.x ? rev1  : rev0;

    const int tid  = threadIdx.x;
    const int lane = tid & 31;
    const int wave = tid >> 5;
    const int ln   = lane & 15, half = lane >> 4;
    const int j0   = (wave & 15) << 4;   // 16-wide hidden-unit slice
    const int m0   = (wave >> 4) << 4;   // batch tile (0 or 16)
    const int j    = j0 + ln;

    for (int i = tid; i < Bn * Hn; i += 1024) hbf[i] = (__bf16)0.f;
    float creg[8];
#pragma unroll
    for (int r = 0; r < 8; ++r) creg[r] = 0.f;
    __syncthreads();

    const int arow = (m0 + ln) * Hn;

    for (int t = 0; t < T; ++t) {
        const int tt = rev ? (T - 1 - t) : t;

        v8f acc[4] = {{}, {}, {}, {}};
#pragma unroll
        for (int kc = 0; kc < 8; ++kc) {
            const int k0 = kc * 32;
            ABfrag a;
#pragma unroll
            for (int v = 0; v < 8; ++v)
                a.u[v] = *(const unsigned*)(&hbf[arow + k0 + frag_koff(v, half)]);
#pragma unroll
            for (int g = 0; g < 4; ++g) {
                ABfrag b;
                const size_t nrow = (size_t)(g * Hn + j) * Hn;
#pragma unroll
                for (int v = 0; v < 8; ++v)
                    b.u[v] = *(const unsigned*)(whh + nrow + k0 + frag_koff(v, half));
                acc[g] = __builtin_amdgcn_wmma_f32_16x16x32_bf16(
                    false, a.v, false, b.v, (short)0, acc[g], false, false);
            }
        }

        // xg add + LSTM nonlinearity (all gates of (m,j) are in this lane)
        float hv[8];
#pragma unroll
        for (int r = 0; r < 8; ++r) {
            const int m = m0 + half * 8 + r;
            const size_t base = xg_has_t ? ((size_t)m * T + tt) * (size_t)Gn
                                         : (size_t)m * Gn;
            const float gi = acc[0][r] + xg[base + j];
            const float gf = acc[1][r] + xg[base + Hn + j];
            const float gg = acc[2][r] + xg[base + 2 * Hn + j];
            const float go = acc[3][r] + xg[base + 3 * Hn + j];
            const float cv = sigf(gf) * creg[r] + sigf(gi) * tanhf(gg);
            creg[r] = cv;
            hv[r] = sigf(go) * tanhf(cv);
        }

        __syncthreads();   // all waves done reading previous h fragments
#pragma unroll
        for (int r = 0; r < 8; ++r) {
            const int m = m0 + half * 8 + r;
            const __bf16 hb = (__bf16)hv[r];
            hbf[m * Hn + j] = hb;
            out[((size_t)m * T + tt) * out_ld + col0 + j] = hb;
        }
        __syncthreads();   // new h visible for next step
    }
}

// ---------------------------------------------------------------------------
// Zero-query attention: q == bq (constant), so one softmax/context per (b,h).
// One block per (b,h); writes head context as bf16 [Bn, En].
// ---------------------------------------------------------------------------
__global__ void attn_kernel(const float* __restrict__ kbuf,
                            const float* __restrict__ vbuf,
                            const float* __restrict__ in_b,
                            __bf16* __restrict__ head_bf) {
    __shared__ float qs[HD];
    __shared__ float sc[TT];
    __shared__ float red[256];

    const int b = blockIdx.x >> 2;
    const int h = blockIdx.x & 3;
    const int tid = threadIdx.x;

    if (tid < HD) qs[tid] = in_b[h * HD + tid];   // bq slice = constant query
    __syncthreads();

    float lmax = -1e30f;
    for (int k = tid; k < TT; k += 256) {
        const float* kr = kbuf + ((size_t)b * TT + k) * En + h * HD;
        float d = 0.f;
        for (int e = 0; e < HD; ++e) d += qs[e] * kr[e];
        d *= 0.08838834764831845f;   // 1/sqrt(128)
        sc[k] = d;
        lmax = fmaxf(lmax, d);
    }
    red[tid] = lmax;
    __syncthreads();
    for (int s = 128; s > 0; s >>= 1) {
        if (tid < s) red[tid] = fmaxf(red[tid], red[tid + s]);
        __syncthreads();
    }
    const float mx = red[0];
    __syncthreads();

    float lsum = 0.f;
    for (int k = tid; k < TT; k += 256) {
        const float e = __expf(sc[k] - mx);
        sc[k] = e;
        lsum += e;
    }
    red[tid] = lsum;
    __syncthreads();
    for (int s = 128; s > 0; s >>= 1) {
        if (tid < s) red[tid] += red[tid + s];
        __syncthreads();
    }
    const float inv = 1.f / red[0];
    __syncthreads();

    if (tid < HD) {
        const float* vb = vbuf + (size_t)b * TT * En + h * HD + tid;
        float o = 0.f;
        for (int k = 0; k < TT; ++k) o += sc[k] * vb[(size_t)k * En];
        head_bf[b * En + h * HD + tid] = (__bf16)(o * inv);
    }
}

// ---------------------------------------------------------------------------
// Host-side orchestration
// ---------------------------------------------------------------------------
extern "C" void kernel_launch(void* const* d_in, const int* in_sizes, int n_in,
                              void* d_out, int out_size, void* d_ws, size_t ws_size,
                              hipStream_t stream) {
    (void)in_sizes; (void)n_in; (void)out_size; (void)ws_size;

    const int*   text      = (const int*)  d_in[0];
    const float* emb       = (const float*)d_in[2];
    const float* encf_wih  = (const float*)d_in[3];
    const float* encf_whh  = (const float*)d_in[4];
    const float* encf_bih  = (const float*)d_in[5];
    const float* encf_bhh  = (const float*)d_in[6];
    const float* encb_wih  = (const float*)d_in[7];
    const float* encb_whh  = (const float*)d_in[8];
    const float* encb_bih  = (const float*)d_in[9];
    const float* encb_bhh  = (const float*)d_in[10];
    const float* attn_in_w = (const float*)d_in[11];
    const float* attn_in_b = (const float*)d_in[12];
    const float* attn_out_w= (const float*)d_in[13];
    const float* attn_out_b= (const float*)d_in[14];
    const float* dec_wih   = (const float*)d_in[15];
    const float* dec_whh   = (const float*)d_in[16];
    const float* dec_bih   = (const float*)d_in[17];
    const float* dec_bhh   = (const float*)d_in[18];
    const float* proj_w    = (const float*)d_in[19];
    const float* proj_b    = (const float*)d_in[20];
    float* outp = (float*)d_out;

    // Workspace carve-out (256B aligned)
    char* ws = (char*)d_ws;
    size_t off = 0;
    auto carve = [&](size_t bytes) -> void* {
        void* p = ws + off;
        off += (bytes + 255) & ~(size_t)255;
        return p;
    };
    __bf16* x_bf     = (__bf16*)carve((size_t)Bn * TT * Hn * 2);
    __bf16* wihf_bf  = (__bf16*)carve((size_t)Gn * Hn * 2);
    __bf16* whhf_bf  = (__bf16*)carve((size_t)Gn * Hn * 2);
    __bf16* wihb_bf  = (__bf16*)carve((size_t)Gn * Hn * 2);
    __bf16* whhb_bf  = (__bf16*)carve((size_t)Gn * Hn * 2);
    __bf16* wkv_bf   = (__bf16*)carve((size_t)2 * En * En * 2);   // Wk then Wv
    __bf16* wout_bf  = (__bf16*)carve((size_t)En * En * 2);
    __bf16* dwih_bf  = (__bf16*)carve((size_t)Gn * En * 2);
    __bf16* dwhh_bf  = (__bf16*)carve((size_t)Gn * Hn * 2);
    __bf16* projw_bf = (__bf16*)carve((size_t)MELn * Hn * 2);
    float*  xg_f     = (float*) carve((size_t)Bn * TT * Gn * 4);
    float*  xg_b     = (float*) carve((size_t)Bn * TT * Gn * 4);
    __bf16* enc_bf   = (__bf16*)carve((size_t)Bn * TT * En * 2);
    float*  kbuf     = (float*) carve((size_t)Bn * TT * En * 4);
    float*  vbuf     = (float*) carve((size_t)Bn * TT * En * 4);
    __bf16* head_bf  = (__bf16*)carve((size_t)Bn * En * 2);
    __bf16* attv_bf  = (__bf16*)carve((size_t)Bn * En * 2);
    float*  dec_xg   = (float*) carve((size_t)Bn * Gn * 4);
    __bf16* dech_bf  = (__bf16*)carve((size_t)Bn * TM * Hn * 2);

    // --- weight conversions to bf16 ---
    auto conv = [&](const float* s, __bf16* d, size_t n) {
        int blk = (int)((n + 255) / 256);
        if (blk > 4096) blk = 4096;
        f32_to_bf16_kernel<<<blk, 256, 0, stream>>>(s, d, n);
    };
    conv(encf_wih, wihf_bf, (size_t)Gn * Hn);
    conv(encf_whh, whhf_bf, (size_t)Gn * Hn);
    conv(encb_wih, wihb_bf, (size_t)Gn * Hn);
    conv(encb_whh, whhb_bf, (size_t)Gn * Hn);
    conv(attn_in_w + (size_t)En * En, wkv_bf, (size_t)2 * En * En); // Wk,Wv
    conv(attn_out_w, wout_bf, (size_t)En * En);
    conv(dec_wih, dwih_bf, (size_t)Gn * En);
    conv(dec_whh, dwhh_bf, (size_t)Gn * Hn);
    conv(proj_w, projw_bf, (size_t)MELn * Hn);

    // --- embedding gather ---
    embed_kernel<<<4096, 256, 0, stream>>>(text, emb, x_bf);

    // --- encoder input gates: xg = x @ Wih^T + (bih+bhh), both directions ---
    {
        const int tiles = (Bn * TT / 16) * (Gn / 16);
        gemm_bf16_wmma<<<tiles / 4, 128, 0, stream>>>(
            x_bf, Hn, wihf_bf, Hn, xg_f, nullptr, Gn,
            encf_bih, encf_bhh, Bn * TT, Gn, Hn);
        gemm_bf16_wmma<<<tiles / 4, 128, 0, stream>>>(
            x_bf, Hn, wihb_bf, Hn, xg_b, nullptr, Gn,
            encb_bih, encb_bhh, Bn * TT, Gn, Hn);
    }

    // --- encoder recurrences (fwd + bwd in one launch, grid.x = 2) ---
    lstm_rec_wmma<<<2, 1024, 0, stream>>>(
        TT, /*xg_has_t=*/1, xg_f, xg_b, whhf_bf, whhb_bf,
        enc_bf, enc_bf, En, /*col fwd=*/0, /*col bwd=*/Hn, /*rev=*/0, 1);

    // --- K / V projections of encoder output ---
    {
        const int tiles = (Bn * TT / 16) * (En / 16);
        gemm_bf16_wmma<<<tiles / 4, 128, 0, stream>>>(
            enc_bf, En, wkv_bf, En, kbuf, nullptr, En,
            attn_in_b + En, nullptr, Bn * TT, En, En);
        gemm_bf16_wmma<<<tiles / 4, 128, 0, stream>>>(
            enc_bf, En, wkv_bf + (size_t)En * En, En, vbuf, nullptr, En,
            attn_in_b + 2 * En, nullptr, Bn * TT, En, En);
    }

    // --- zero-query attention -> per-batch context heads ---
    attn_kernel<<<Bn * NHEAD, 256, 0, stream>>>(kbuf, vbuf, attn_in_b, head_bf);

    // --- attention output projection: att = head @ Wout^T + bout (bf16) ---
    gemm_bf16_wmma<<<(Bn / 16) * (En / 16) / 4, 128, 0, stream>>>(
        head_bf, En, wout_bf, En, nullptr, attv_bf, En,
        attn_out_b, nullptr, Bn, En, En);

    // --- decoder input gates (time-invariant!): dec_xg = att @ Wih^T + b ---
    gemm_bf16_wmma<<<(Bn / 16) * (Gn / 16) / 4, 128, 0, stream>>>(
        attv_bf, En, dwih_bf, En, dec_xg, nullptr, Gn,
        dec_bih, dec_bhh, Bn, Gn, En);

    // --- decoder recurrence (1000 steps) ---
    lstm_rec_wmma<<<1, 1024, 0, stream>>>(
        TM, /*xg_has_t=*/0, dec_xg, dec_xg, dwhh_bf, dwhh_bf,
        dech_bf, dech_bf, Hn, 0, 0, 0, 0);

    // --- mel projection straight into d_out [B*TM, 80] ---
    gemm_bf16_wmma<<<(Bn * TM / 16) * (MELn / 16) / 4, 128, 0, stream>>>(
        dech_bf, Hn, projw_bf, Hn, outp, nullptr, MELn,
        proj_b, nullptr, Bn * TM, MELn, Hn);
}
